// DCADecoderBlock_87840671138119
// MI455X (gfx1250) — matrix operand order repaired and checked by reading, not compiled, not run on hardware
//
#include <hip/hip_runtime.h>

// ---------------------------------------------------------------------------
// DCA decoder block for MI455X (gfx1250, wave32, WMMA bf16 16x16x32 + TDM)
// ---------------------------------------------------------------------------

#define DEV static __device__ __forceinline__

constexpr int   Dm   = 1024;
constexpr int   Sm   = 2048;
constexpr int   Bm   = 2;
constexpr int   Lm   = 4;
constexpr int   Hm   = 16;
constexpr int   HDm  = 64;
constexpr int   MLPm = 4096;
constexpr float EPSF = 1e-5f;

typedef __attribute__((ext_vector_type(16))) __bf16 v16bf;
typedef __attribute__((ext_vector_type(8)))  __bf16 bf8v;
typedef __attribute__((ext_vector_type(8)))  float  v8f;

DEV v16bf mkfrag(bf8v lo, bf8v hi) {
    union U { v16bf v; bf8v h[2]; } u;
    u.h[0] = lo; u.h[1] = hi;
    return u.v;
}

DEV v8f wmma_bf16(v16bf a, v16bf b, v8f c) {
    // D = A(16x32 bf16) * B(32x16 bf16) + C(16x16 f32)
    return __builtin_amdgcn_wmma_f32_16x16x32_bf16(
        /*neg_a=*/false, a, /*neg_b=*/false, b,
        /*c_mod=*/(short)0, c, /*reuse_a=*/false, /*reuse_b=*/false);
}

// ---------------------------------------------------------------------------
// Tensor Data Mover: async 2-D bf16 tile load global -> LDS.
// Descriptor layout per CDNA5 ISA (D# groups 0/1).
//   - data_size = 1 (2 bytes / element)
//   - pad_enable inserts pad_amount DWORDs every 2^(pad_interval+1) DWORDs,
//     reproducing the padded LDS row stride used by the fragment loaders.
// Issue from one wave; completion via TENSORcnt.
// ---------------------------------------------------------------------------
#if __has_builtin(__builtin_amdgcn_tensor_load_to_lds)
#define HAVE_TDM 1
#else
#define HAVE_TDM 0
#endif

#if HAVE_TDM
typedef __attribute__((ext_vector_type(4))) unsigned int u32x4;
typedef __attribute__((ext_vector_type(8))) int          i32x8;
typedef __attribute__((ext_vector_type(4))) int          i32x4;

DEV void tdm_load_2d(const void* gptr, const void* ldsPtr,
                     unsigned int stride_elems,   // tensor dim0 stride (elems)
                     unsigned int tile_d0,        // tile width  (elems)
                     unsigned int tile_d1,        // tile height (rows)
                     unsigned int pad_interval,   // code: pad every 2^(c+1) DW
                     unsigned int pad_amount) {   // code: (c+1) DWORDs pad
    unsigned long long ga = (unsigned long long)(uintptr_t)gptr;
    unsigned int       la = (unsigned int)(uintptr_t)ldsPtr;  // low32 = LDS off
    const unsigned int td1 = 1u << 20;            // generous bound, no clip

    u32x4 g0;
    g0[0] = 1u;                                          // count=1, user mode
    g0[1] = la;                                          // lds_addr
    g0[2] = (unsigned int)(ga & 0xFFFFFFFFu);            // global_addr[31:0]
    g0[3] = (unsigned int)((ga >> 32) & 0x01FFFFFFu)     // global_addr[56:32]
          | (2u << 30);                                  // type = 2 (image)

    i32x8 g1;
    g1[0] = (int)((1u << 16)                             // data_size = 2B
                | (1u << 20)                             // pad_enable
                | (pad_interval << 22)
                | (pad_amount << 25));
    g1[1] = (int)((stride_elems & 0xFFFFu) << 16);       // tensor_dim0[15:0]
    g1[2] = (int)(((stride_elems >> 16) & 0xFFFFu)       // tensor_dim0[31:16]
                | ((td1 & 0xFFFFu) << 16));              // tensor_dim1[15:0]
    g1[3] = (int)(((td1 >> 16) & 0xFFFFu)                // tensor_dim1[31:16]
                | (tile_d0 << 16));                      // tile_dim0
    g1[4] = (int)(tile_d1 & 0xFFFFu);                    // tile_dim1 (dim2=0)
    g1[5] = (int)stride_elems;                           // dim0_stride[31:0]
    g1[6] = 0;                                           // stride[47:32], d1s
    g1[7] = 0;

    i32x4 z = {0, 0, 0, 0};
#if defined(__clang_major__) && (__clang_major__ >= 23)
    i32x8 z8 = {0, 0, 0, 0, 0, 0, 0, 0};
    __builtin_amdgcn_tensor_load_to_lds(g0, g1, z, z, z8, 0);
#else
    __builtin_amdgcn_tensor_load_to_lds(g0, g1, z, z, 0);
#endif
}
DEV void tdm_wait() { __builtin_amdgcn_s_wait_tensorcnt(0); }
#endif

// ---------------------------------------------------------------------------
// f32 -> bf16 conversion (grid-stride)
// ---------------------------------------------------------------------------
__global__ void k_cvt_bf16(const float* __restrict__ src,
                           __bf16* __restrict__ dst, int n) {
    int i = blockIdx.x * blockDim.x + threadIdx.x;
    int stride = gridDim.x * blockDim.x;
    for (; i < n; i += stride) dst[i] = (__bf16)src[i];
}

// ---------------------------------------------------------------------------
// normsum[b,s,:] = sum_l normalize(stacked[b,l,s,:])    (LN core, no affine)
// ---------------------------------------------------------------------------
__global__ __launch_bounds__(256) void k_normsum(
        const float* __restrict__ stacked, float* __restrict__ normsum) {
    const int r   = blockIdx.x;          // b*S + s
    const int b   = r >> 11;
    const int s   = r & (Sm - 1);
    const int tid = threadIdx.x;
    __shared__ float red[256];

    float acc[4] = {0.f, 0.f, 0.f, 0.f};
    for (int l = 0; l < Lm; ++l) {
        const float* row = stacked + ((size_t)(b * Lm + l) * Sm + s) * Dm;
        float x[4]; float sum = 0.f;
        #pragma unroll
        for (int j = 0; j < 4; ++j) { x[j] = row[tid + 256 * j]; sum += x[j]; }
        red[tid] = sum; __syncthreads();
        for (int o = 128; o > 0; o >>= 1) {
            if (tid < o) red[tid] += red[tid + o];
            __syncthreads();
        }
        const float mu = red[0] * (1.0f / Dm);
        __syncthreads();
        float vs = 0.f;
        #pragma unroll
        for (int j = 0; j < 4; ++j) { float d = x[j] - mu; vs += d * d; }
        red[tid] = vs; __syncthreads();
        for (int o = 128; o > 0; o >>= 1) {
            if (tid < o) red[tid] += red[tid + o];
            __syncthreads();
        }
        const float rinv = rsqrtf(red[0] * (1.0f / Dm) + EPSF);
        __syncthreads();
        #pragma unroll
        for (int j = 0; j < 4; ++j) acc[j] += (x[j] - mu) * rinv;
    }
    #pragma unroll
    for (int j = 0; j < 4; ++j)
        normsum[(size_t)r * Dm + tid + 256 * j] = acc[j];
}

// ---------------------------------------------------------------------------
// GRN GEMM:  out[b,s,e] = sum_l relu(stacked[b,l,s,:] @ W[:,e] + bias[e])
//                        + g[e]*normsum[b,s,e] + L*be[e]       (stored bf16)
// WG tile 128(M) x 64(N), 8 waves each own a 32x32 subtile (2x2 WMMA).
// A tile staged by TDM; W tile transposed by VALU path (overlaps TDM).
// ---------------------------------------------------------------------------
__global__ __launch_bounds__(256) void k_grn_gemm(
        const __bf16* __restrict__ A,        // stacked bf16 [B,L,S,D]
        const __bf16* __restrict__ W,        // [D,D] row-major (K,N)
        const float*  __restrict__ bias,
        const float*  __restrict__ gamma,
        const float*  __restrict__ beta,
        const float*  __restrict__ normsum,  // [B*S, D]
        __bf16*       __restrict__ out) {    // [B*S, D] bf16
    __shared__ __align__(16) __bf16 ldsA[128][40];   // 32-elem rows + 8 pad
    __shared__ __align__(16) __bf16 ldsWt[64][40];   // transposed: [N][K]

    const int tid  = threadIdx.x;
    const int w    = tid >> 5;
    const int lane = tid & 31;
    const int half = lane >> 4;
    const int ln16 = lane & 15;
    const int m0   = blockIdx.y * 128;      // global row base (b*S+s)
    const int n0   = blockIdx.x * 64;
    const int bIdx = m0 >> 11;
    const int sB   = m0 & (Sm - 1);
    const int mw   = (w & 3) * 32;
    const int nw   = (w >> 2) * 32;

    v8f sumacc[2][2] = {};
    for (int l = 0; l < Lm; ++l) {
        v8f acc[2][2] = {};
        const __bf16* Abase = A + ((size_t)(bIdx * Lm + l) * Sm + sB) * Dm;
        for (int k0 = 0; k0 < Dm; k0 += 32) {
            __syncthreads();
#if HAVE_TDM
            if (w == 0)   // async: 128 rows x 32 elems, pad 4 DW per 16 DW
                tdm_load_2d(Abase + k0, &ldsA[0][0], Dm, 32, 128, 3, 3);
#else
            #pragma unroll
            for (int it = 0; it < 2; ++it) {
                int c = tid + 256 * it;
                int row = c >> 2, kg = (c & 3) * 8;
                *(bf8v*)&ldsA[row][kg] =
                    *(const bf8v*)&Abase[(size_t)row * Dm + k0 + kg];
            }
#endif
            {   // W tile 32x64 -> transposed (overlaps TDM transfer)
                int kr = tid >> 3, ng = (tid & 7) * 8;
                bf8v v = *(const bf8v*)&W[(size_t)(k0 + kr) * Dm + n0 + ng];
                #pragma unroll
                for (int j = 0; j < 8; ++j) ldsWt[ng + j][kr] = v[j];
            }
#if HAVE_TDM
            if (w == 0) tdm_wait();
#endif
            __syncthreads();
            #pragma unroll
            for (int mi = 0; mi < 2; ++mi) {
                const int ar = mw + 16 * mi + ln16;
                v16bf af = mkfrag(*(const bf8v*)&ldsA[ar][8 * half],
                                  *(const bf8v*)&ldsA[ar][16 + 8 * half]);
                #pragma unroll
                for (int ni = 0; ni < 2; ++ni) {
                    const int bc = nw + 16 * ni + ln16;
                    v16bf bf = mkfrag(*(const bf8v*)&ldsWt[bc][8 * half],
                                      *(const bf8v*)&ldsWt[bc][16 + 8 * half]);
                    acc[mi][ni] = wmma_bf16(af, bf, acc[mi][ni]);
                }
            }
        }
        // relu(+bias), accumulate over l
        #pragma unroll
        for (int ni = 0; ni < 2; ++ni) {
            const float bc = bias[n0 + nw + 16 * ni + ln16];
            #pragma unroll
            for (int mi = 0; mi < 2; ++mi)
                #pragma unroll
                for (int i = 0; i < 8; ++i)
                    sumacc[mi][ni][i] += fmaxf(acc[mi][ni][i] + bc, 0.f);
        }
    }
    // epilogue: + g*normsum + L*be, store bf16
    #pragma unroll
    for (int ni = 0; ni < 2; ++ni) {
        const int col = n0 + nw + 16 * ni + ln16;
        const float g = gamma[col], be = beta[col];
        #pragma unroll
        for (int mi = 0; mi < 2; ++mi)
            #pragma unroll
            for (int i = 0; i < 8; ++i) {
                const int row = m0 + mw + 16 * mi + i + 8 * half;
                float v = sumacc[mi][ni][i] +
                          g * normsum[(size_t)row * Dm + col] + 4.0f * be;
                out[(size_t)row * Dm + col] = (__bf16)v;
            }
    }
}

// ---------------------------------------------------------------------------
// Generic bf16 WMMA GEMM: C = A[M,K] @ W[K,N] (+bias)
// mode 0: store f32; mode 1: GELU -> bf16
// WG tile 128x128, 8 waves, wave tile 64x32 (4x2 WMMA). A tile via TDM.
// ---------------------------------------------------------------------------
__global__ __launch_bounds__(256) void k_gemm(
        const __bf16* __restrict__ A, const __bf16* __restrict__ W,
        const float* __restrict__ bias,
        float* __restrict__ outF, __bf16* __restrict__ outB,
        int M, int N, int K, int mode) {
    __shared__ __align__(16) __bf16 ldsA[128][40];
    __shared__ __align__(16) __bf16 ldsWt[128][40];

    const int tid  = threadIdx.x;
    const int w    = tid >> 5;
    const int lane = tid & 31;
    const int half = lane >> 4;
    const int ln16 = lane & 15;
    const int m0   = blockIdx.y * 128;
    const int n0   = blockIdx.x * 128;
    const int mw   = (w & 1) * 64;
    const int nw   = (w >> 1) * 32;

    v8f acc[4][2] = {};
    for (int k0 = 0; k0 < K; k0 += 32) {
        __syncthreads();
#if HAVE_TDM
        if (w == 0)
            tdm_load_2d(&A[(size_t)m0 * K + k0], &ldsA[0][0], K, 32, 128, 3, 3);
#else
        #pragma unroll
        for (int it = 0; it < 2; ++it) {
            int c = tid + 256 * it;
            int row = c >> 2, kg = (c & 3) * 8;
            *(bf8v*)&ldsA[row][kg] =
                *(const bf8v*)&A[(size_t)(m0 + row) * K + k0 + kg];
        }
#endif
        {   // W tile 32x128 -> transposed (overlaps TDM transfer)
            int kr = tid >> 3;
            #pragma unroll
            for (int it = 0; it < 2; ++it) {
                int ng = ((tid & 7) + 8 * it) * 8;
                bf8v v = *(const bf8v*)&W[(size_t)(k0 + kr) * N + n0 + ng];
                #pragma unroll
                for (int j = 0; j < 8; ++j) ldsWt[ng + j][kr] = v[j];
            }
        }
#if HAVE_TDM
        if (w == 0) tdm_wait();
#endif
        __syncthreads();
        #pragma unroll
        for (int mi = 0; mi < 4; ++mi) {
            const int ar = mw + 16 * mi + ln16;
            v16bf af = mkfrag(*(const bf8v*)&ldsA[ar][8 * half],
                              *(const bf8v*)&ldsA[ar][16 + 8 * half]);
            #pragma unroll
            for (int ni = 0; ni < 2; ++ni) {
                const int bc = nw + 16 * ni + ln16;
                v16bf bf = mkfrag(*(const bf8v*)&ldsWt[bc][8 * half],
                                  *(const bf8v*)&ldsWt[bc][16 + 8 * half]);
                acc[mi][ni] = wmma_bf16(af, bf, acc[mi][ni]);
            }
        }
    }
    #pragma unroll
    for (int ni = 0; ni < 2; ++ni) {
        const int col  = n0 + nw + 16 * ni + ln16;
        const float bc = bias[col];
        #pragma unroll
        for (int mi = 0; mi < 4; ++mi)
            #pragma unroll
            for (int i = 0; i < 8; ++i) {
                const int row = m0 + mw + 16 * mi + i + 8 * half;
                float v = acc[mi][ni][i] + bc;
                if (mode == 1) {
                    v = 0.5f * v * (1.0f + erff(v * 0.70710678118654752f));
                    outB[(size_t)row * N + col] = (__bf16)v;
                } else {
                    outF[(size_t)row * N + col] = v;
                }
            }
    }
}

// ---------------------------------------------------------------------------
// Flash attention, one WG per (q-tile of 128, b*H). 8 waves x 16 q rows.
// K tile via TDM; V tile transposed manually; P via per-wave LDS scratch.
// ---------------------------------------------------------------------------
__global__ __launch_bounds__(256) void k_attn(
        const __bf16* __restrict__ qb, const __bf16* __restrict__ kb,
        const __bf16* __restrict__ vb, __bf16* __restrict__ ob) {
    __shared__ __align__(16) __bf16 ldsK[32][72];       // 64-elem rows + 8 pad
    __shared__ __align__(16) __bf16 ldsVt[64][40];      // [dim][k-row]
    __shared__ __align__(16) __bf16 ldsP[8][16][40];    // per-wave P (16x32)

    const int tid  = threadIdx.x;
    const int w    = tid >> 5;
    const int lane = tid & 31;
    const int half = lane >> 4;
    const int ln16 = lane & 15;
    const int bh   = blockIdx.y;
    const int b    = bh >> 4;
    const int h    = bh & 15;
    const int q0   = blockIdx.x * 128;
    const size_t dOff = (size_t)h * HDm;
    const float scale = 0.125f;            // HD^-0.5

    // preload q fragments (16 rows x 64 dims per wave)
    const int qrow = q0 + w * 16 + ln16;
    const __bf16* qrp = qb + ((size_t)(b * Sm) + qrow) * Dm + dOff;
    v16bf qf[2];
    #pragma unroll
    for (int f = 0; f < 2; ++f)
        qf[f] = mkfrag(*(const bf8v*)&qrp[32 * f + 8 * half],
                       *(const bf8v*)&qrp[32 * f + 16 + 8 * half]);

    float m_run[8], l_run[8];
    #pragma unroll
    for (int i = 0; i < 8; ++i) { m_run[i] = -1e30f; l_run[i] = 0.f; }
    v8f accO[4] = {};

    for (int kb0 = 0; kb0 < Sm; kb0 += 32) {
        __syncthreads();
#if HAVE_TDM
        if (w == 0)   // 32 rows x 64 elems, pad 4 DW per 32 DW
            tdm_load_2d(kb + ((size_t)(b * Sm) + kb0) * Dm + dOff,
                        &ldsK[0][0], Dm, 64, 32, 4, 3);
#endif
        {   // V tile 32x64 -> transposed [dim][k-row] (overlaps TDM)
            int kr = tid >> 3, dg = (tid & 7) * 8;
#if !HAVE_TDM
            const __bf16* krow = kb + ((size_t)(b * Sm) + kb0 + kr) * Dm + dOff;
            *(bf8v*)&ldsK[kr][dg] = *(const bf8v*)&krow[dg];
#endif
            const __bf16* vrow = vb + ((size_t)(b * Sm) + kb0 + kr) * Dm + dOff;
            bf8v vv = *(const bf8v*)&vrow[dg];
            #pragma unroll
            for (int j = 0; j < 8; ++j) ldsVt[dg + j][kr] = vv[j];
        }
#if HAVE_TDM
        if (w == 0) tdm_wait();
#endif
        __syncthreads();

        // scores: 16 q rows x 32 k cols (two 16x16 tiles, K=64 over 2 WMMAs)
        v8f s0 = {}, s1 = {};
        #pragma unroll
        for (int f = 0; f < 2; ++f) {
            v16bf b0 = mkfrag(*(const bf8v*)&ldsK[ln16][32 * f + 8 * half],
                              *(const bf8v*)&ldsK[ln16][32 * f + 16 + 8 * half]);
            v16bf b1 = mkfrag(*(const bf8v*)&ldsK[16 + ln16][32 * f + 8 * half],
                              *(const bf8v*)&ldsK[16 + ln16][32 * f + 16 + 8 * half]);
            s0 = wmma_bf16(qf[f], b0, s0);
            s1 = wmma_bf16(qf[f], b1, s1);
        }

        // online softmax over the 32 new columns
        #pragma unroll
        for (int i = 0; i < 8; ++i) {
            float a0 = s0[i] * scale, a1 = s1[i] * scale;
            float mx = fmaxf(a0, a1);
            #pragma unroll
            for (int off = 1; off < 16; off <<= 1)
                mx = fmaxf(mx, __shfl_xor(mx, off, 16));
            float mnew = fmaxf(m_run[i], mx);
            float c  = __expf(m_run[i] - mnew);
            float p0 = __expf(a0 - mnew);
            float p1 = __expf(a1 - mnew);
            float rs = p0 + p1;
            #pragma unroll
            for (int off = 1; off < 16; off <<= 1)
                rs += __shfl_xor(rs, off, 16);
            l_run[i] = l_run[i] * c + rs;
            m_run[i] = mnew;
            #pragma unroll
            for (int t = 0; t < 4; ++t) accO[t][i] *= c;
            const int pr = i + 8 * half;
            ldsP[w][pr][ln16]      = (__bf16)p0;
            ldsP[w][pr][16 + ln16] = (__bf16)p1;
        }

        // P (16x32) @ V (32x64)
        v16bf pf = mkfrag(*(const bf8v*)&ldsP[w][ln16][8 * half],
                          *(const bf8v*)&ldsP[w][ln16][16 + 8 * half]);
        #pragma unroll
        for (int t = 0; t < 4; ++t) {
            v16bf vf = mkfrag(*(const bf8v*)&ldsVt[16 * t + ln16][8 * half],
                              *(const bf8v*)&ldsVt[16 * t + ln16][16 + 8 * half]);
            accO[t] = wmma_bf16(pf, vf, accO[t]);
        }
    }

    // finalize: O /= l, write bf16 [B,S,D]
    #pragma unroll
    for (int i = 0; i < 8; ++i) {
        const float inv = 1.0f / l_run[i];
        const int orow = q0 + w * 16 + i + 8 * half;
        #pragma unroll
        for (int t = 0; t < 4; ++t)
            ob[((size_t)(b * Sm) + orow) * Dm + dOff + 16 * t + ln16] =
                (__bf16)(accO[t][i] * inv);
    }
}

// ---------------------------------------------------------------------------
// outF = resin + LN(hsrc)*g + b ; optional bf16 copy. One block per row.
// ---------------------------------------------------------------------------
__global__ __launch_bounds__(256) void k_ln_res(
        const float* __restrict__ hsrc, const float* __restrict__ resin,
        const float* __restrict__ g, const float* __restrict__ bvec,
        float* __restrict__ outF, __bf16* __restrict__ outB) {
    const int r   = blockIdx.x;
    const int tid = threadIdx.x;
    __shared__ float red[256];
    const float* row = hsrc + (size_t)r * Dm;

    float x[4]; float sum = 0.f;
    #pragma unroll
    for (int j = 0; j < 4; ++j) { x[j] = row[tid + 256 * j]; sum += x[j]; }
    red[tid] = sum; __syncthreads();
    for (int o = 128; o > 0; o >>= 1) {
        if (tid < o) red[tid] += red[tid + o];
        __syncthreads();
    }
    const float mu = red[0] * (1.0f / Dm);
    __syncthreads();
    float vs = 0.f;
    #pragma unroll
    for (int j = 0; j < 4; ++j) { float d = x[j] - mu; vs += d * d; }
    red[tid] = vs; __syncthreads();
    for (int o = 128; o > 0; o >>= 1) {
        if (tid < o) red[tid] += red[tid + o];
        __syncthreads();
    }
    const float rinv = rsqrtf(red[0] * (1.0f / Dm) + EPSF);
    #pragma unroll
    for (int j = 0; j < 4; ++j) {
        const int col = tid + 256 * j;
        float v = resin[(size_t)r * Dm + col] +
                  (x[j] - mu) * rinv * g[col] + bvec[col];
        outF[(size_t)r * Dm + col] = v;
        if (outB) outB[(size_t)r * Dm + col] = (__bf16)v;
    }
}

// ---------------------------------------------------------------------------
// Host launcher
// ---------------------------------------------------------------------------
extern "C" void kernel_launch(void* const* d_in, const int* in_sizes, int n_in,
                              void* d_out, int out_size, void* d_ws, size_t ws_size,
                              hipStream_t stream) {
    (void)in_sizes; (void)n_in; (void)out_size; (void)ws_size;

    const float* x        = (const float*)d_in[0];
    const float* stacked  = (const float*)d_in[1];
    const float* gW[3]    = {(const float*)d_in[2], (const float*)d_in[6],  (const float*)d_in[10]};
    const float* gb[3]    = {(const float*)d_in[3], (const float*)d_in[7],  (const float*)d_in[11]};
    const float* gg[3]    = {(const float*)d_in[4], (const float*)d_in[8],  (const float*)d_in[12]};
    const float* gbe[3]   = {(const float*)d_in[5], (const float*)d_in[9],  (const float*)d_in[13]};
    const float* attW     = (const float*)d_in[14];
    const float* attb     = (const float*)d_in[15];
    const float* anormg   = (const float*)d_in[16];
    const float* anormb   = (const float*)d_in[17];
    const float* W1       = (const float*)d_in[18];
    const float* b1       = (const float*)d_in[19];
    const float* W2       = (const float*)d_in[20];
    const float* b2       = (const float*)d_in[21];
    const float* fnormg   = (const float*)d_in[22];
    const float* fnormb   = (const float*)d_in[23];

    const int BS  = Bm * Sm;              // 4096
    const int BLS = Bm * Lm * Sm;         // 16384

    size_t off = 0;
    auto carve = [&](size_t bytes) -> void* {
        void* p = (char*)d_ws + off;
        off += (bytes + 255) & ~(size_t)255;
        return p;
    };
    __bf16* sBF     = (__bf16*)carve((size_t)BLS * Dm * 2);
    __bf16* wq      = (__bf16*)carve((size_t)Dm * Dm * 2);
    __bf16* wk      = (__bf16*)carve((size_t)Dm * Dm * 2);
    __bf16* wv      = (__bf16*)carve((size_t)Dm * Dm * 2);
    __bf16* wo      = (__bf16*)carve((size_t)Dm * Dm * 2);
    __bf16* w1      = (__bf16*)carve((size_t)Dm * MLPm * 2);
    __bf16* w2      = (__bf16*)carve((size_t)MLPm * Dm * 2);
    float*  normsum = (float*) carve((size_t)BS * Dm * 4);
    __bf16* qbf     = (__bf16*)carve((size_t)BS * Dm * 2);
    __bf16* kbf     = (__bf16*)carve((size_t)BS * Dm * 2);
    __bf16* vbf     = (__bf16*)carve((size_t)BS * Dm * 2);
    __bf16* attnbf  = (__bf16*)carve((size_t)BS * Dm * 2);
    float*  tF      = (float*) carve((size_t)BS * Dm * 4);
    float*  res1    = (float*) carve((size_t)BS * Dm * 4);
    __bf16* res1b   = (__bf16*)carve((size_t)BS * Dm * 2);
    __bf16* hbf     = (__bf16*)carve((size_t)BS * MLPm * 2);
    float*  h2      = (float*) carve((size_t)BS * Dm * 4);

    auto cvt = [&](const float* s, __bf16* d, int n) {
        int blocks = (n + 255) / 256;
        if (blocks > 16384) blocks = 16384;
        k_cvt_bf16<<<blocks, 256, 0, stream>>>(s, d, n);
    };
    cvt(stacked, sBF, BLS * Dm);
    cvt(gW[0], wq, Dm * Dm);
    cvt(gW[1], wk, Dm * Dm);
    cvt(gW[2], wv, Dm * Dm);
    cvt(attW,  wo, Dm * Dm);
    cvt(W1,    w1, Dm * MLPm);
    cvt(W2,    w2, MLPm * Dm);

    k_normsum<<<BS, 256, 0, stream>>>(stacked, normsum);

    // GRN q/k/v
    {
        dim3 grid(Dm / 64, BS / 128);
        k_grn_gemm<<<grid, 256, 0, stream>>>(sBF, wq, gb[0], gg[0], gbe[0], normsum, qbf);
        k_grn_gemm<<<grid, 256, 0, stream>>>(sBF, wk, gb[1], gg[1], gbe[1], normsum, kbf);
        k_grn_gemm<<<grid, 256, 0, stream>>>(sBF, wv, gb[2], gg[2], gbe[2], normsum, vbf);
    }

    // flash attention
    {
        dim3 grid(Sm / 128, Bm * Hm);
        k_attn<<<grid, 256, 0, stream>>>(qbf, kbf, vbf, attnbf);
    }

    // attn_out projection (f32 out, +bias)
    {
        dim3 grid(Dm / 128, BS / 128);
        k_gemm<<<grid, 256, 0, stream>>>(attnbf, wo, attb, tF, nullptr,
                                         BS, Dm, Dm, 0);
    }
    // res1 = x + LN(tF)   (f32 + bf16 copy)
    k_ln_res<<<BS, 256, 0, stream>>>(tF, x, anormg, anormb, res1, res1b);

    // FFN1: gelu(res1 @ W1 + b1) -> bf16
    {
        dim3 grid(MLPm / 128, BS / 128);
        k_gemm<<<grid, 256, 0, stream>>>(res1b, w1, b1, nullptr, hbf,
                                         BS, MLPm, Dm, 1);
    }
    // FFN2: hbf @ W2 + b2 -> f32
    {
        dim3 grid(Dm / 128, BS / 128);
        k_gemm<<<grid, 256, 0, stream>>>(hbf, w2, b2, h2, nullptr,
                                         BS, Dm, MLPm, 0);
    }
    // out = res1 + LN(h2)
    k_ln_res<<<BS, 256, 0, stream>>>(h2, res1, fnormg, fnormb,
                                     (float*)d_out, nullptr);
}
